// CeptaTransformerBlock_53300544143618
// MI455X (gfx1250) — compile-verified, matrix operand order checked by reading, not run on hardware
//
#include <hip/hip_runtime.h>
#include <hip/hip_bf16.h>

// ---------------------------------------------------------------------------
// Types for CDNA5 WMMA
// ---------------------------------------------------------------------------
typedef __attribute__((ext_vector_type(16))) __bf16 v16bf;
typedef __attribute__((ext_vector_type(8)))  float  v8f;

union Frag {
    v16bf v;
    uint4 q[2];
};

// ---------------------------------------------------------------------------
// Device helpers
// ---------------------------------------------------------------------------
__device__ __forceinline__ unsigned short f2bf(float f) {
    unsigned int u = __float_as_uint(f);
    u += 0x7FFFu + ((u >> 16) & 1u);     // round-to-nearest-even
    return (unsigned short)(u >> 16);
}

__device__ __forceinline__ float gelu_f(float x) {
    const float c0 = 0.7978845608028654f;   // sqrt(2/pi)
    const float c1 = 0.044715f;
    return 0.5f * x * (1.0f + tanhf(c0 * (x + c1 * x * x * x)));
}

__device__ __forceinline__ float sigmoid_f(float x) {
    return 1.0f / (1.0f + expf(-x));
}

// ---------------------------------------------------------------------------
// Weight prep: B (K x N, f32, row-major) -> B^T (Np x Kp, bf16, row-major),
// zero-padded. In this layout the (k,k+1) bf16 pair needed by the WMMA
// B-fragment is a naturally-aligned little-endian u32 — no packing in GEMM.
// ---------------------------------------------------------------------------
__global__ void cvt_bt_kernel(const float* __restrict__ src,
                              unsigned short* __restrict__ dst,
                              int K, int N, int Kp, int Np) {
    int i = blockIdx.x * blockDim.x + threadIdx.x;
    if (i >= Np * Kp) return;
    int n = i / Kp, k = i % Kp;
    float v = (k < K && n < N) ? src[(size_t)k * N + n] : 0.0f;
    dst[i] = f2bf(v);
}

// ---------------------------------------------------------------------------
// RMSNorm: one 256-thread block per row, D = 1024 (4 elems/thread), bf16 out
// ---------------------------------------------------------------------------
__global__ __launch_bounds__(256)
void rmsnorm_kernel(const float* __restrict__ x, const float* __restrict__ w,
                    unsigned short* __restrict__ out, int D) {
    __shared__ float red[256];
    const int row = blockIdx.x;
    const int tid = threadIdx.x;
    const float4 xv = *(const float4*)(x + (size_t)row * D + tid * 4);
    float s = xv.x * xv.x + xv.y * xv.y + xv.z * xv.z + xv.w * xv.w;
    red[tid] = s;
    __syncthreads();
    for (int st = 128; st > 0; st >>= 1) {
        if (tid < st) red[tid] += red[tid + st];
        __syncthreads();
    }
    const float inv = rsqrtf(red[0] / (float)D + 1e-6f);
    const float4 wv = *(const float4*)(w + tid * 4);
    unsigned short* o = out + (size_t)row * D + tid * 4;
    o[0] = f2bf(xv.x * wv.x * inv);
    o[1] = f2bf(xv.y * wv.y * inv);
    o[2] = f2bf(xv.z * wv.z * inv);
    o[3] = f2bf(xv.w * wv.w * inv);
}

// ---------------------------------------------------------------------------
// WMMA GEMM:  C[M,N] = A[M,K](bf16, row-major) * B (given as B^T: N x K bf16
// row-major), f32 accumulate.
//
// Block tile 64x128, BK=32, 8 waves (2x4), each wave 32x32 via 4 accumulators.
// Both A and B^T tiles stage as straight 16B copies; fragments load as two
// ds_load_b128 each (CDNA5 16-bit A/B VGPR layout: lanes 0..15 hold
// k=0..7,16..23; lanes 16..31 hold k=8..15,24..31; two bf16 per VGPR).
//
// Requires: M % 64 == 0, N % 128 == 0, K % 32 == 0 (small matrices zero-padded).
//
// Epilogue modes:
//   0: outF = C
//   1: outB = bf16(gelu(C))
//   2: outB = bf16(C + addbuf)
//   3: outF = C + addbuf
//   4: outB = bf16(C)
// ---------------------------------------------------------------------------
#define LDSTRIDE 40   // 32 k-values + 8 pad (u16 units)

template<int MODE>
__global__ __launch_bounds__(256)
void gemm_bf16_kernel(const unsigned short* __restrict__ A,
                      const unsigned short* __restrict__ Bt,
                      const float* __restrict__ addbuf,
                      float* __restrict__ outF,
                      unsigned short* __restrict__ outB,
                      int M, int N, int K) {
    __shared__ alignas(16) unsigned short As[64 * LDSTRIDE];    // (m, k)
    __shared__ alignas(16) unsigned short Bst[128 * LDSTRIDE];  // (n, k)

    const int tid  = threadIdx.x;
    const int wave = tid >> 5;
    const int lane = tid & 31;
    const int wr   = wave >> 2;          // wave row 0..1  (32 rows each)
    const int wc   = wave & 3;           // wave col 0..3  (32 cols each)
    const int m0   = blockIdx.y * 64;
    const int n0   = blockIdx.x * 128;

    // A staging: 64x32 tile, one uint4 per thread
    const int arow = tid >> 2;           // 0..63
    const int acol = (tid & 3) * 8;      // 0,8,16,24
    // B staging: 128x32 tile (n-major), two uint4 per thread
    const int bnr = tid >> 1;            // 0..127
    const int bkc = (tid & 1) * 16;      // 0 or 16

    const int hh  = lane >> 4;           // wave half
    const int l15 = lane & 15;

    v8f acc[2][2] = {};

    for (int k0 = 0; k0 < K; k0 += 32) {
        // ---- stage A ----
        const uint4 av = *(const uint4*)(A + (size_t)(m0 + arow) * K + k0 + acol);
        *(uint4*)(As + arow * LDSTRIDE + acol) = av;
        // ---- stage B^T (already n-major, pairs pre-packed) ----
        const unsigned short* gb = Bt + (size_t)(n0 + bnr) * K + k0 + bkc;
        const uint4 bv0 = *(const uint4*)(gb);
        const uint4 bv1 = *(const uint4*)(gb + 8);
        *(uint4*)(Bst + bnr * LDSTRIDE + bkc)     = bv0;
        *(uint4*)(Bst + bnr * LDSTRIDE + bkc + 8) = bv1;
        if (k0 + 32 < K) {
            __builtin_prefetch(A + (size_t)(m0 + arow) * K + (k0 + 32) + acol, 0, 3);
            __builtin_prefetch(gb + 32, 0, 3);
        }
        __syncthreads();

        // ---- fragments: two ds_load_b128 each ----
        Frag a[2], b[2];
        #pragma unroll
        for (int r = 0; r < 2; ++r) {
            const unsigned short* pa = As + (wr * 32 + r * 16 + l15) * LDSTRIDE;
            a[r].q[0] = *(const uint4*)(pa + 8 * hh);
            a[r].q[1] = *(const uint4*)(pa + 16 + 8 * hh);
        }
        #pragma unroll
        for (int e = 0; e < 2; ++e) {
            const unsigned short* pb = Bst + (wc * 32 + e * 16 + l15) * LDSTRIDE;
            b[e].q[0] = *(const uint4*)(pb + 8 * hh);
            b[e].q[1] = *(const uint4*)(pb + 16 + 8 * hh);
        }

        #pragma unroll
        for (int r = 0; r < 2; ++r)
            #pragma unroll
            for (int e = 0; e < 2; ++e)
                acc[r][e] = __builtin_amdgcn_wmma_f32_16x16x32_bf16(
                    false, a[r].v, false, b[e].v, (short)0, acc[r][e], false, false);

        __syncthreads();
    }

    // ---- epilogue: C/D layout — VGPR i: lanes 0..15 -> M=i, 16..31 -> M=i+8
    #pragma unroll
    for (int r = 0; r < 2; ++r) {
        #pragma unroll
        for (int e = 0; e < 2; ++e) {
            const int ncol = n0 + wc * 32 + e * 16 + l15;
            #pragma unroll
            for (int i = 0; i < 8; ++i) {
                const int mrow = m0 + wr * 32 + r * 16 + i + 8 * hh;
                const size_t idx = (size_t)mrow * N + ncol;
                float val = acc[r][e][i];
                if (MODE == 1)      { outB[idx] = f2bf(gelu_f(val)); }
                else if (MODE == 2) { outB[idx] = f2bf(val + addbuf[idx]); }
                else if (MODE == 3) { outF[idx] = val + addbuf[idx]; }
                else if (MODE == 4) { outB[idx] = f2bf(val); }
                else                { outF[idx] = val; }
            }
        }
    }
}

// ---------------------------------------------------------------------------
// t = pre_path + sum_a gelu(U[:, a*P + p])
// ---------------------------------------------------------------------------
__global__ void alpha_reduce_kernel(const float* __restrict__ U,
                                    const float* __restrict__ pre,
                                    float* __restrict__ tout,
                                    int MT, int P, int A) {
    int idx = blockIdx.x * blockDim.x + threadIdx.x;
    if (idx >= MT * P) return;
    int m = idx / P, p = idx % P;
    float s = pre[idx];
    for (int a = 0; a < A; ++a)
        s += gelu_f(U[(size_t)m * (A * P) + a * P + p]);
    tout[idx] = s;
}

// ---------------------------------------------------------------------------
// Gated scan: h_t = sigmoid(flogit_t) * h_{t-1} + t_t, per (b,p) channel.
// One thread per channel; loads coalesced over p. Emits h as bf16.
// ---------------------------------------------------------------------------
__global__ void scan_kernel(const float* __restrict__ tin,
                            const float* __restrict__ flogit,
                            unsigned short* __restrict__ hb,
                            int B, int T, int P) {
    int ch = blockIdx.x * blockDim.x + threadIdx.x;
    if (ch >= B * P) return;
    int b = ch / P, p = ch % P;
    float h = 0.0f;
    size_t base = (size_t)b * T * P + p;
    for (int t = 0; t < T; ++t) {
        size_t i = base + (size_t)t * P;
        float f = sigmoid_f(flogit[i]);
        h = f * h + tin[i];
        hb[i] = f2bf(h);
    }
}

// ---------------------------------------------------------------------------
// Host side
// ---------------------------------------------------------------------------
extern "C" void kernel_launch(void* const* d_in, const int* in_sizes, int n_in,
                              void* d_out, int out_size, void* d_ws, size_t ws_size,
                              hipStream_t stream) {
    (void)in_sizes; (void)n_in; (void)out_size; (void)ws_size;

    constexpr int B = 2, T = 2048, D = 1024, P = 512, ALPHA = 4, PR = 64;
    constexpr int PRP = 128;             // PR zero-padded to block-tile N
    constexpr int HID = 16 * D;
    constexpr int MT = B * T;            // 4096 token rows

    const float* x       = (const float*)d_in[0];
    const float* w_norm1 = (const float*)d_in[1];
    const float* w_toP   = (const float*)d_in[2];
    const float* w_emb   = (const float*)d_in[3];
    const float* w_f     = (const float*)d_in[4];
    const float* b_r     = (const float*)d_in[5];
    const float* c_r     = (const float*)d_in[6];
    const float* w_fromP = (const float*)d_in[7];
    const float* w_norm2 = (const float*)d_in[8];
    const float* w_fc1   = (const float*)d_in[9];
    const float* w_fc2   = (const float*)d_in[10];
    float* out = (float*)d_out;

    // workspace carve-up (256B aligned)
    size_t off = 0;
    auto alloc = [&](size_t bytes) -> char* {
        char* p = (char*)d_ws + off;
        off += (bytes + 255) & ~(size_t)255;
        return p;
    };
    // weights stored transposed (N x K, bf16)
    unsigned short* wtoP_t   = (unsigned short*)alloc((size_t)P * D * 2);
    unsigned short* wemb_t   = (unsigned short*)alloc((size_t)(ALPHA * P) * D * 2);
    unsigned short* wf_t     = (unsigned short*)alloc((size_t)P * D * 2);
    unsigned short* br_t     = (unsigned short*)alloc((size_t)PRP * P * 2);
    unsigned short* cr_t     = (unsigned short*)alloc((size_t)P * PRP * 2);
    unsigned short* wfromP_t = (unsigned short*)alloc((size_t)D * P * 2);
    unsigned short* wfc1_t   = (unsigned short*)alloc((size_t)HID * D * 2);
    unsigned short* wfc2_t   = (unsigned short*)alloc((size_t)D * HID * 2);
    // activations
    unsigned short* h1b      = (unsigned short*)alloc((size_t)MT * D * 2);
    float*          preb     = (float*)alloc((size_t)MT * P * 4);
    float*          Ubuf     = (float*)alloc((size_t)MT * ALPHA * P * 4);
    float*          Fraw     = (float*)alloc((size_t)MT * P * 4);
    float*          tbuf     = (float*)alloc((size_t)MT * P * 4);
    unsigned short* hb       = (unsigned short*)alloc((size_t)MT * P * 2);
    unsigned short* Gb       = (unsigned short*)alloc((size_t)MT * PRP * 2);
    unsigned short* ttb      = (unsigned short*)alloc((size_t)MT * P * 2);
    float*          x2       = (float*)alloc((size_t)MT * D * 4);
    unsigned short* h2b      = (unsigned short*)alloc((size_t)MT * D * 2);
    unsigned short* abuf     = (unsigned short*)alloc((size_t)MT * HID * 2);

    // 1) weights -> bf16, transposed (+ zero-pad PR dim to 128)
    auto cvtT = [&](const float* src, unsigned short* dst, int K, int N, int Kp, int Np) {
        int n = Np * Kp;
        cvt_bt_kernel<<<(n + 255) / 256, 256, 0, stream>>>(src, dst, K, N, Kp, Np);
    };
    cvtT(w_toP,   wtoP_t,   D,   P,         D,   P);
    cvtT(w_emb,   wemb_t,   D,   ALPHA * P, D,   ALPHA * P);
    cvtT(w_f,     wf_t,     D,   P,         D,   P);
    cvtT(b_r,     br_t,     P,   PR,        P,   PRP);
    cvtT(c_r,     cr_t,     PR,  P,         PRP, P);
    cvtT(w_fromP, wfromP_t, P,   D,         P,   D);
    cvtT(w_fc1,   wfc1_t,   D,   HID,       D,   HID);
    cvtT(w_fc2,   wfc2_t,   HID, D,         HID, D);

    // 2) h1 = rmsnorm(x, w_norm1) -> bf16
    rmsnorm_kernel<<<MT, 256, 0, stream>>>(x, w_norm1, h1b, D);

    auto grid = [](int M, int N) { return dim3((unsigned)(N / 128), (unsigned)(M / 64)); };

    // 3) context projections off h1
    gemm_bf16_kernel<0><<<grid(MT, P), 256, 0, stream>>>(
        h1b, wtoP_t, nullptr, preb, nullptr, MT, P, D);
    gemm_bf16_kernel<0><<<grid(MT, ALPHA * P), 256, 0, stream>>>(
        h1b, wemb_t, nullptr, Ubuf, nullptr, MT, ALPHA * P, D);
    gemm_bf16_kernel<0><<<grid(MT, P), 256, 0, stream>>>(
        h1b, wf_t, nullptr, Fraw, nullptr, MT, P, D);

    // 4) t = pre + sum_a gelu(U)
    alpha_reduce_kernel<<<(MT * P + 255) / 256, 256, 0, stream>>>(
        Ubuf, preb, tbuf, MT, P, ALPHA);

    // 5) gated scan -> h (bf16)
    scan_kernel<<<(B * P + 255) / 256, 256, 0, stream>>>(tbuf, Fraw, hb, B, T, P);

    // 6) low-rank correction: G = h @ b_r (bf16, padded N=128);
    //    t_tilde = t + G @ c_r (bf16) with K=128 (zero cols contribute 0)
    gemm_bf16_kernel<4><<<grid(MT, PRP), 256, 0, stream>>>(
        hb, br_t, nullptr, nullptr, Gb, MT, PRP, P);
    gemm_bf16_kernel<2><<<grid(MT, P), 256, 0, stream>>>(
        Gb, cr_t, tbuf, nullptr, ttb, MT, P, PRP);

    // 7) x2 = x + t_tilde @ w_fromP ; h2 = rmsnorm(x2) bf16
    gemm_bf16_kernel<3><<<grid(MT, D), 256, 0, stream>>>(
        ttb, wfromP_t, x, x2, nullptr, MT, D, P);
    rmsnorm_kernel<<<MT, 256, 0, stream>>>(x2, w_norm2, h2b, D);

    // 8) MLP: a = gelu(h2 @ w_fc1) bf16 ; out = x2 + a @ w_fc2
    gemm_bf16_kernel<1><<<grid(MT, HID), 256, 0, stream>>>(
        h2b, wfc1_t, nullptr, nullptr, abuf, MT, HID, D);
    gemm_bf16_kernel<3><<<grid(MT, D), 256, 0, stream>>>(
        abuf, wfc2_t, x2, out, nullptr, MT, D, HID);
}
